// Glm4MoeMLPStack_63221918597199
// MI455X (gfx1250) — matrix operand order repaired and checked by reading, compiled, not run on hardware
//
#include <hip/hip_runtime.h>
#include <hip/hip_bf16.h>

// Problem constants (fixed by the reference)
#define NEXP 32
#define HDIM 2048
#define IDIM 1408
#define TTOK 32768

// Tiling: 128x128 block tile, 512 threads = 16 waves, each wave 32x32
#define BM 128
#define BN 128
#define BK 32
#define LDK 40   // padded K-stride (elements) in LDS: 80B rows, conflict-free
#define NTHR 512

typedef __attribute__((ext_vector_type(16))) __bf16 v16bf;
typedef __attribute__((ext_vector_type(8)))  __bf16 v8bf;
typedef __attribute__((ext_vector_type(4)))  __bf16 v4bf;
typedef __attribute__((ext_vector_type(8)))  float  v8f;
typedef __attribute__((ext_vector_type(4)))  unsigned int v4u;
typedef __attribute__((ext_vector_type(8)))  int v8i;
typedef __attribute__((ext_vector_type(4)))  int v4i;

// ---- Tensor Data Mover availability (device pass only) --------------------
#if defined(__HIP_DEVICE_COMPILE__)
#if __has_builtin(__builtin_amdgcn_tensor_load_to_lds)
#define HAVE_TDM 1
#else
#define HAVE_TDM 0
#endif
#else
#define HAVE_TDM 0
#endif

static __device__ __forceinline__ v8f wmma_bf16(v16bf a, v16bf b, v8f c) {
  return __builtin_amdgcn_wmma_f32_16x16x32_bf16(false, a, false, b, (short)0, c,
                                                 false, false);
}

// A fragment (16x32 bf16, MxK): lane holds row M=lane&15; kg=lane>>4.
// element j -> K=kg*16+j. LDS: A[m*LDK + k].
static __device__ __forceinline__ v16bf load_a_frag(const __bf16* As, int mBase,
                                                    int lane) {
  const int m  = mBase + (lane & 15);
  const int kg = lane >> 4;
  const __bf16* row = As + m * LDK + kg * 16;
  v8bf lo = *reinterpret_cast<const v8bf*>(row);
  v8bf hi = *reinterpret_cast<const v8bf*>(row + 8);
  v16bf a;
#pragma unroll
  for (int j = 0; j < 8; ++j) { a[j] = lo[j]; a[j + 8] = hi[j]; }
  return a;
}

// B fragment (32x16 bf16, KxN): lane holds col N=lane&15, K=(lane>>4)*16+j.
// LDS stores B transposed: Bt[n*LDK + k] -> K contiguous.
static __device__ __forceinline__ v16bf load_b_frag(const __bf16* Bt, int nBase,
                                                    int lane) {
  const int n  = nBase + (lane & 15);
  const int kg = lane >> 4;
  const __bf16* row = Bt + n * LDK + kg * 16;
  v8bf lo = *reinterpret_cast<const v8bf*>(row);
  v8bf hi = *reinterpret_cast<const v8bf*>(row + 8);
  v16bf b;
#pragma unroll
  for (int j = 0; j < 8; ++j) { b[j] = lo[j]; b[j + 8] = hi[j]; }
  return b;
}

static __device__ __forceinline__ int expert_of_row(const int* __restrict__ gs,
                                                    int row) {
  int e = 0, acc = 0;
#pragma unroll 1
  for (int i = 0; i < NEXP; ++i) {
    int g = gs[i];
    if (row < acc + g) { e = i; break; }
    acc += g;
  }
  return e;
}

#if HAVE_TDM
// TDM: load a [128 x 32] bf16 tile from a row-major [*, IDIM] tensor into LDS.
// pad_interval=3 (pad after every 16 DWORDs = one 32-elem row),
// pad_amount=3 (+4 DWORDs = 8 elems) -> LDS row stride 40 elems == LDK.
static __device__ __forceinline__ void tdm_load_hidden_tile(const __bf16* gptr,
                                                            void* lds_dst) {
  unsigned long long ga = (unsigned long long)(uintptr_t)gptr;
  unsigned lo = (unsigned)(uintptr_t)lds_dst;  // low 32 bits = LDS byte offset
  v4u g0;
  g0[0] = 1u;                                   // count=1, user D#
  g0[1] = lo;                                   // lds_addr
  g0[2] = (unsigned)ga;                         // global_addr[31:0]
  g0[3] = ((unsigned)(ga >> 32) & 0x01FFFFFFu)  // global_addr[56:32]
          | (2u << 30);                         // type=2 ("image")
  v8i g1;
  g1[0] = (int)((1u << 16)        // data_size = 2 bytes
                | (1u << 20)      // pad_enable
                | (3u << 22)      // pad_interval: 16 DWORDs
                | (3u << 25));    // pad_amount: 4 DWORDs
  g1[1] = (int)(((unsigned)IDIM & 0xffffu) << 16);          // tensor_dim0 lo
  g1[2] = (int)(((unsigned)IDIM >> 16) |
                (((unsigned)TTOK & 0xffffu) << 16));        // dim0 hi | dim1 lo
  g1[3] = (int)(((unsigned)TTOK >> 16) | (32u << 16));      // dim1 hi | tile_dim0
  g1[4] = 128;                                              // tile_dim1=128
  g1[5] = IDIM;                                             // dim0_stride lo
  g1[6] = 0;
  g1[7] = 0;
  v4i z = {0, 0, 0, 0};
#if __clang_major__ >= 23
  v8i z8 = {0, 0, 0, 0, 0, 0, 0, 0};
  __builtin_amdgcn_tensor_load_to_lds(g0, g1, z, z, z8, 0);
#else
  __builtin_amdgcn_tensor_load_to_lds(g0, g1, z, z, 0);
#endif
}

static __device__ __forceinline__ void wait_tensorcnt0() {
#if __has_builtin(__builtin_amdgcn_s_wait_tensorcnt)
  __builtin_amdgcn_s_wait_tensorcnt(0);
#else
  asm volatile("s_wait_tensorcnt 0x0" ::: "memory");
#endif
}
#endif  // HAVE_TDM

// -------- Pass 1: hidden = silu(x@Wg) * (x@Wu), bf16 to workspace ----------
__global__ __launch_bounds__(NTHR) void moe_gateup_kernel(
    const float* __restrict__ x, const float* __restrict__ wg,
    const float* __restrict__ wu, const int* __restrict__ gs,
    unsigned short* __restrict__ hidden_raw) {
  __shared__ __bf16 As[2][BM * LDK];
  __shared__ __bf16 Bg[2][BN * LDK];
  __shared__ __bf16 Bu[2][BN * LDK];
  __bf16* hidden = reinterpret_cast<__bf16*>(hidden_raw);

  const int r0 = blockIdx.y * BM;
  const int c0 = blockIdx.x * BN;
  const int e  = expert_of_row(gs, r0);
  const float* wge = wg + (size_t)e * HDIM * IDIM;
  const float* wue = wu + (size_t)e * HDIM * IDIM;

  const int tid  = threadIdx.x;
  const int lane = tid & 31;
  const int wave = tid >> 5;          // 0..15, grid 4(M) x 4(N)
  const int wm   = (wave >> 2) * 32;
  const int wn   = (wave & 3) * 32;

  // Loop-invariant per-thread 32-bit offsets (element units)
  int aoff[2], alds[2], boff[2], blds[2];
#pragma unroll
  for (int it = 0; it < 2; ++it) {
    int idx = tid + it * NTHR;                 // 0..1023 float4 chunks
    int r = idx >> 3, c4 = (idx & 7) * 4;      // A: 128 x 8 float4
    aoff[it] = (r0 + r) * HDIM + c4;
    alds[it] = r * LDK + c4;
    int k = idx >> 5, n4 = (idx & 31) * 4;     // B: 32 x 32 float4
    boff[it] = k * IDIM + c0 + n4;
    blds[it] = n4 * LDK + k;
  }

  v8f accG[2][2], accU[2][2];
#pragma unroll
  for (int i = 0; i < 2; ++i)
#pragma unroll
    for (int t = 0; t < 2; ++t)
#pragma unroll
      for (int r = 0; r < 8; ++r) { accG[i][t][r] = 0.f; accU[i][t][r] = 0.f; }

  // Prologue: stage K-block 0 into buffer 0
#pragma unroll
  for (int it = 0; it < 2; ++it) {
    float4 v = *reinterpret_cast<const float4*>(&x[aoff[it]]);
    v4bf b = {(__bf16)v.x, (__bf16)v.y, (__bf16)v.z, (__bf16)v.w};
    *reinterpret_cast<v4bf*>(&As[0][alds[it]]) = b;
  }
#pragma unroll
  for (int it = 0; it < 2; ++it) {
    float4 g = *reinterpret_cast<const float4*>(&wge[boff[it]]);
    float4 u = *reinterpret_cast<const float4*>(&wue[boff[it]]);
    Bg[0][blds[it] + 0 * LDK] = (__bf16)g.x;
    Bg[0][blds[it] + 1 * LDK] = (__bf16)g.y;
    Bg[0][blds[it] + 2 * LDK] = (__bf16)g.z;
    Bg[0][blds[it] + 3 * LDK] = (__bf16)g.w;
    Bu[0][blds[it] + 0 * LDK] = (__bf16)u.x;
    Bu[0][blds[it] + 1 * LDK] = (__bf16)u.y;
    Bu[0][blds[it] + 2 * LDK] = (__bf16)u.z;
    Bu[0][blds[it] + 3 * LDK] = (__bf16)u.w;
  }
  __syncthreads();

  const int NK = HDIM / BK;
  for (int s = 0; s < NK; ++s) {
    const int cur = s & 1, nxt = cur ^ 1;
    const int kA = (s + 1) * BK;          // uniform scalar K advance (A)
    const int kB = (s + 1) * BK * IDIM;   // uniform scalar K advance (B)
    const bool more = (s + 1) < NK;

    // (a) issue global loads for the next K-block, convert to bf16 in regs
    v4bf sa[2], sg[2], su[2];
    if (more) {
#pragma unroll
      for (int it = 0; it < 2; ++it) {
        float4 v = *reinterpret_cast<const float4*>(&x[aoff[it] + kA]);
        v4bf b = {(__bf16)v.x, (__bf16)v.y, (__bf16)v.z, (__bf16)v.w};
        sa[it] = b;
      }
#pragma unroll
      for (int it = 0; it < 2; ++it) {
        float4 g = *reinterpret_cast<const float4*>(&wge[boff[it] + kB]);
        float4 u = *reinterpret_cast<const float4*>(&wue[boff[it] + kB]);
        v4bf gb = {(__bf16)g.x, (__bf16)g.y, (__bf16)g.z, (__bf16)g.w};
        v4bf ub = {(__bf16)u.x, (__bf16)u.y, (__bf16)u.z, (__bf16)u.w};
        sg[it] = gb;
        su[it] = ub;
      }
    }

    // (b) compute on current buffer (ds_loads issued before staging ds_stores)
    v16bf af[2];
#pragma unroll
    for (int i = 0; i < 2; ++i) af[i] = load_a_frag(As[cur], wm + i * 16, lane);
#pragma unroll
    for (int t = 0; t < 2; ++t) {
      v16bf bgf = load_b_frag(Bg[cur], wn + t * 16, lane);
      v16bf buf = load_b_frag(Bu[cur], wn + t * 16, lane);
#pragma unroll
      for (int i = 0; i < 2; ++i) {
        accG[i][t] = wmma_bf16(af[i], bgf, accG[i][t]);
        accU[i][t] = wmma_bf16(af[i], buf, accU[i][t]);
      }
    }

    // (c) store staged regs into the other buffer
    if (more) {
#pragma unroll
      for (int it = 0; it < 2; ++it)
        *reinterpret_cast<v4bf*>(&As[nxt][alds[it]]) = sa[it];
#pragma unroll
      for (int it = 0; it < 2; ++it) {
        Bg[nxt][blds[it] + 0 * LDK] = sg[it][0];
        Bg[nxt][blds[it] + 1 * LDK] = sg[it][1];
        Bg[nxt][blds[it] + 2 * LDK] = sg[it][2];
        Bg[nxt][blds[it] + 3 * LDK] = sg[it][3];
        Bu[nxt][blds[it] + 0 * LDK] = su[it][0];
        Bu[nxt][blds[it] + 1 * LDK] = su[it][1];
        Bu[nxt][blds[it] + 2 * LDK] = su[it][2];
        Bu[nxt][blds[it] + 3 * LDK] = su[it][3];
      }
    }
    __syncthreads();
  }

  // SwiGLU epilogue: fast rcp instead of IEEE division
  const int kg = lane >> 4;
#pragma unroll
  for (int i = 0; i < 2; ++i)
#pragma unroll
    for (int t = 0; t < 2; ++t)
#pragma unroll
      for (int r = 0; r < 8; ++r) {
        int m = wm + i * 16 + kg * 8 + r;
        int n = wn + t * 16 + (lane & 15);
        float g = accG[i][t][r], u = accU[i][t][r];
        float h = g * __builtin_amdgcn_rcpf(1.0f + __expf(-g)) * u;
        hidden[(size_t)(r0 + m) * IDIM + c0 + n] = (__bf16)h;
      }
}

// -------- Pass 2: out = hidden @ Wd, fp32 out ------------------------------
__global__ __launch_bounds__(NTHR) void moe_down_kernel(
    const unsigned short* __restrict__ hidden_raw,
    const float* __restrict__ wd, const int* __restrict__ gs,
    float* __restrict__ out) {
  __shared__ __bf16 As[2][BM * LDK];
  __shared__ __bf16 Bd[2][BN * LDK];
  const __bf16* hidden = reinterpret_cast<const __bf16*>(hidden_raw);

  const int r0 = blockIdx.y * BM;
  const int c0 = blockIdx.x * BN;
  const int e  = expert_of_row(gs, r0);
  const float* wde = wd + (size_t)e * IDIM * HDIM;

  const int tid  = threadIdx.x;
  const int lane = tid & 31;
  const int wave = tid >> 5;
  const int wm   = (wave >> 2) * 32;
  const int wn   = (wave & 3) * 32;

  int boff[2], blds[2];
#pragma unroll
  for (int it = 0; it < 2; ++it) {
    int idx = tid + it * NTHR;
    int k = idx >> 5, n4 = (idx & 31) * 4;
    boff[it] = k * HDIM + c0 + n4;
    blds[it] = n4 * LDK + k;
  }
#if !HAVE_TDM
  // one v8bf chunk per thread covers the 128x32 bf16 tile
  int hoff, hlds;
  {
    int r = tid >> 2, c8 = (tid & 3) * 8;
    hoff = (r0 + r) * IDIM + c8;
    hlds = r * LDK + c8;
  }
#endif

  v8f acc[2][2];
#pragma unroll
  for (int i = 0; i < 2; ++i)
#pragma unroll
    for (int t = 0; t < 2; ++t)
#pragma unroll
      for (int r = 0; r < 8; ++r) acc[i][t][r] = 0.f;

  // Prologue: stage K-block 0 into buffer 0
#if HAVE_TDM
  if (wave == 0)
    tdm_load_hidden_tile(hidden + (size_t)r0 * IDIM, &As[0][0]);
#else
  {
    v8bf v = *reinterpret_cast<const v8bf*>(&hidden[hoff]);
    *reinterpret_cast<v8bf*>(&As[0][hlds]) = v;
  }
#endif
#pragma unroll
  for (int it = 0; it < 2; ++it) {
    float4 v = *reinterpret_cast<const float4*>(&wde[boff[it]]);
    Bd[0][blds[it] + 0 * LDK] = (__bf16)v.x;
    Bd[0][blds[it] + 1 * LDK] = (__bf16)v.y;
    Bd[0][blds[it] + 2 * LDK] = (__bf16)v.z;
    Bd[0][blds[it] + 3 * LDK] = (__bf16)v.w;
  }
#if HAVE_TDM
  if (wave == 0) wait_tensorcnt0();
#endif
  __syncthreads();

  const int NK = IDIM / BK;  // 44
  for (int s = 0; s < NK; ++s) {
    const int cur = s & 1, nxt = cur ^ 1;
    const int kA = (s + 1) * BK;          // uniform K advance (hidden)
    const int kB = (s + 1) * BK * HDIM;   // uniform K advance (w_down)
    const bool more = (s + 1) < NK;

    // (a) next A via Tensor Data Mover (wave 0), next B via regs
#if HAVE_TDM
    if (more && wave == 0)
      tdm_load_hidden_tile(hidden + (size_t)r0 * IDIM + kA, &As[nxt][0]);
#else
    v8bf sa;
    if (more) sa = *reinterpret_cast<const v8bf*>(&hidden[hoff + kA]);
#endif
    v4bf sb[2];
    if (more) {
#pragma unroll
      for (int it = 0; it < 2; ++it) {
        float4 v = *reinterpret_cast<const float4*>(&wde[boff[it] + kB]);
        v4bf b = {(__bf16)v.x, (__bf16)v.y, (__bf16)v.z, (__bf16)v.w};
        sb[it] = b;
      }
    }

    // (b) compute on current buffer
    v16bf af[2];
#pragma unroll
    for (int i = 0; i < 2; ++i) af[i] = load_a_frag(As[cur], wm + i * 16, lane);
#pragma unroll
    for (int t = 0; t < 2; ++t) {
      v16bf bf = load_b_frag(Bd[cur], wn + t * 16, lane);
#pragma unroll
      for (int i = 0; i < 2; ++i) acc[i][t] = wmma_bf16(af[i], bf, acc[i][t]);
    }

    // (c) store staged B (and A fallback) into the other buffer
    if (more) {
#pragma unroll
      for (int it = 0; it < 2; ++it) {
        Bd[nxt][blds[it] + 0 * LDK] = sb[it][0];
        Bd[nxt][blds[it] + 1 * LDK] = sb[it][1];
        Bd[nxt][blds[it] + 2 * LDK] = sb[it][2];
        Bd[nxt][blds[it] + 3 * LDK] = sb[it][3];
      }
#if !HAVE_TDM
      *reinterpret_cast<v8bf*>(&As[nxt][hlds]) = sa;
#endif
    }
#if HAVE_TDM
    if (wave == 0) wait_tensorcnt0();
#endif
    __syncthreads();
  }

  const int kg = lane >> 4;
#pragma unroll
  for (int i = 0; i < 2; ++i)
#pragma unroll
    for (int t = 0; t < 2; ++t)
#pragma unroll
      for (int r = 0; r < 8; ++r) {
        int m = wm + i * 16 + kg * 8 + r;
        int n = wn + t * 16 + (lane & 15);
        out[(size_t)(r0 + m) * HDIM + c0 + n] = acc[i][t][r];
      }
}

extern "C" void kernel_launch(void* const* d_in, const int* in_sizes, int n_in,
                              void* d_out, int out_size, void* d_ws,
                              size_t ws_size, hipStream_t stream) {
  const float* x  = (const float*)d_in[0];
  const float* wg = (const float*)d_in[1];
  const float* wu = (const float*)d_in[2];
  const float* wd = (const float*)d_in[3];
  const int*   gs = (const int*)d_in[4];
  float* out = (float*)d_out;
  unsigned short* hidden = (unsigned short*)d_ws;  // T*I bf16 = 88 MB scratch

  dim3 blk(NTHR);
  dim3 gA(IDIM / BN, TTOK / BM);  // (11, 256)
  dim3 gB(HDIM / BN, TTOK / BM);  // (16, 256)
  moe_gateup_kernel<<<gA, blk, 0, stream>>>(x, wg, wu, gs, hidden);
  moe_down_kernel<<<gB, blk, 0, stream>>>(hidden, wd, gs, out);
}
